// Attention_45724221833993
// MI455X (gfx1250) — matrix operand order repaired and checked by reading, not compiled, not run on hardware
//
#include <hip/hip_runtime.h>
#include <math.h>

// ---------------------------------------------------------------------------
// Problem constants (match reference)
// ---------------------------------------------------------------------------
#define BB      4
#define NN      16384          // H*W
#define CC      128
#define HEADS   2
#define DH      64             // CC / HEADS
#define SRR     8
#define NKV     256            // (128/8)*(128/8) kv tokens per image
#define MTOT    (BB*NN)        // 65536 flat tokens
#define XHEADS  8
#define XD      16             // CC / XHEADS

typedef __bf16 bf16_t;
typedef __attribute__((ext_vector_type(16))) __bf16 bf16x16;
typedef __attribute__((ext_vector_type(8)))  float  f32x8;
typedef __attribute__((ext_vector_type(4)))  int    i32x4;

#define DEVINL __device__ __forceinline__

// Optional gfx1250 async-to-LDS path (guarded; falls back to plain LDS copy)
#if defined(__gfx1250__) && __has_builtin(__builtin_amdgcn_global_load_async_to_lds_b128) && __has_builtin(__builtin_amdgcn_s_wait_asynccnt)
#define USE_ASYNC_LDS 1
typedef __attribute__((address_space(1))) i32x4 glob_i32x4;
typedef __attribute__((address_space(3))) i32x4 lds_i32x4;
DEVINL lds_i32x4* to_lds(void* p) {
  // generic shared pointer: low 32 bits are the LDS byte offset (ISA 10.2)
  return (lds_i32x4*)(unsigned int)(unsigned long long)p;
}
DEVINL glob_i32x4* to_glob(const void* p) {
  return (glob_i32x4*)(unsigned long long)p;
}
#else
#define USE_ASYNC_LDS 0
#endif

DEVINL f32x8 zero8() {
  f32x8 z;
  #pragma unroll
  for (int i = 0; i < 8; ++i) z[i] = 0.0f;
  return z;
}

DEVINL f32x8 wmma_bf16(bf16x16 a, bf16x16 b, f32x8 c) {
  // D = A(16x32 bf16) x B(32x16 bf16) + C(16x16 f32)
  return __builtin_amdgcn_wmma_f32_16x16x32_bf16(
      false, a, false, b, (short)0, c, false, false);
}

// A fragment (16x32): rows 0..15 from row-major src (leading dim ld),
// K columns kbase..kbase+31.  Lane L<16 holds row L, K = {0..7,16..23}+kh;
// lane L>=16 holds row L-16, K = {8..15,24..31} (kh=8).
DEVINL bf16x16 load_a(const bf16_t* __restrict__ src, int ld, int kbase, int lane) {
  const int row = lane & 15;
  const int kh  = (lane >> 4) << 3;            // 0 or 8
  const bf16_t* p = src + (size_t)row * ld + kbase + kh;
  bf16x16 a;
  #pragma unroll
  for (int j = 0; j < 8; ++j) { a[j] = p[j]; a[8 + j] = p[16 + j]; }
  return a;
}

// B fragment (32x16) for computing x @ W^T: B[k][n] = W[n][k].
// Wn0 points at W row n0 (row-major, leading dim ld).  Lanes 0..15 cover
// K = {0..7,16..23}, lanes 16..31 cover K = {8..15,24..31} (+kbase).
DEVINL bf16x16 load_b(const bf16_t* __restrict__ Wn0, int ld, int kbase, int lane) {
  const int n  = lane & 15;
  const int kh = (lane >> 4) << 3;             // 0 or 8
  const bf16_t* p = Wn0 + (size_t)n * ld + kbase + kh;
  bf16x16 b;
  #pragma unroll
  for (int j = 0; j < 8; ++j) { b[j] = p[j]; b[8 + j] = p[16 + j]; }
  return b;
}

// C/D tile layout helpers: VGPR i <-> row (i or i+8), lane&15 <-> col.
DEVINL int crow(int lane, int i) { return ((lane >> 4) << 3) + i; }
DEVINL int ccol(int lane)        { return lane & 15; }

DEVINL float half16_reduce_add(float v) {
  #pragma unroll
  for (int m = 1; m <= 8; m <<= 1) v += __shfl_xor(v, m, 32);
  return v;                                     // sums lanes 0-15 / 16-31 separately
}
DEVINL float half16_reduce_max(float v) {
  #pragma unroll
  for (int m = 1; m <= 8; m <<= 1) v = fmaxf(v, __shfl_xor(v, m, 32));
  return v;
}

// ---------------------------------------------------------------------------
// Elementwise conversion / repack kernels
// ---------------------------------------------------------------------------
__global__ void cvt_f32_bf16(const float* __restrict__ s, bf16_t* __restrict__ d, int n) {
  int i = blockIdx.x * blockDim.x + threadIdx.x;
  if (i < n) d[i] = (bf16_t)s[i];
}

// sr_w (co, ci, ky, kx) -> (co, ky*8+kx, ci) bf16 so im2col A-frags are contiguous
__global__ void repack_srw(const float* __restrict__ s, bf16_t* __restrict__ d) {
  int i = blockIdx.x * blockDim.x + threadIdx.x;
  if (i >= CC * CC * SRR * SRR) return;
  int co = i >> 13;          // /8192
  int r  = i & 8191;
  int p  = r >> 7;           // ky*8+kx
  int ci = r & 127;
  d[i] = (bf16_t)s[((size_t)co * CC + ci) * (SRR * SRR) + p];
}

// ---------------------------------------------------------------------------
// Generic GEMM:  Ob[M x (NT*16)] = A[M x (KS*32)] @ W^T + bias   (bf16 out)
// block = 128 threads = 4 waves, each wave owns a 16-row tile.
// B-fragments are batch-loaded (chunks of 8) before the WMMAs so the
// compiler emits one load clause + graduated s_wait_loadcnt.
// ---------------------------------------------------------------------------
template <int NT, int KS>
__global__ void __launch_bounds__(128)
gemm_bias_bf16(const bf16_t* __restrict__ A, const bf16_t* __restrict__ W,
               const float* __restrict__ bias, bf16_t* __restrict__ Ob,
               int M, int ldo) {
  const int lane = threadIdx.x & 31, wave = threadIdx.x >> 5;
  const int m0 = (blockIdx.x * 4 + wave) * 16;
  if (m0 >= M) return;
  const int K = KS * 32;
  const bf16_t* Arow = A + (size_t)m0 * K;

  f32x8 acc[NT];
  #pragma unroll
  for (int nt = 0; nt < NT; ++nt) acc[nt] = zero8();

  #pragma unroll
  for (int ks = 0; ks < KS; ++ks) {
    bf16x16 a = load_a(Arow, K, ks * 32, lane);
    __builtin_prefetch(Arow + (size_t)16 * K, 0, 1);
    #pragma unroll
    for (int c0 = 0; c0 < NT; c0 += 8) {
      bf16x16 bfr[8];
      #pragma unroll
      for (int j = 0; j < 8 && c0 + j < NT; ++j)
        bfr[j] = load_b(W + (size_t)((c0 + j) * 16) * K, K, ks * 32, lane);
      #pragma unroll
      for (int j = 0; j < 8 && c0 + j < NT; ++j)
        acc[c0 + j] = wmma_bf16(a, bfr[j], acc[c0 + j]);
    }
  }
  #pragma unroll
  for (int nt = 0; nt < NT; ++nt) {
    const int col = nt * 16 + ccol(lane);
    const float bv = bias[col];
    #pragma unroll
    for (int i = 0; i < 8; ++i) {
      const int r = m0 + crow(lane, i);
      Ob[(size_t)r * ldo + col] = (bf16_t)(acc[nt][i] + bv);
    }
  }
}

// ---------------------------------------------------------------------------
// SR conv as im2col GEMM:  out[1024 x 128] f32,  K = 64*128 = 8192
// row m = (b, oy, ox);  A[m, p*128+ci] = x[b, (oy*8+ky)*128 + ox*8+kx, ci]
// ---------------------------------------------------------------------------
__global__ void __launch_bounds__(128)
conv_kernel(const bf16_t* __restrict__ xb, const bf16_t* __restrict__ srw,
            const float* __restrict__ srb, float* __restrict__ out) {
  const int lane = threadIdx.x & 31, wave = threadIdx.x >> 5;
  const int m0  = (blockIdx.x * 4 + wave) * 16;
  const int row = m0 + (lane & 15);
  const int b   = row >> 8, rem = row & 255, oy = rem >> 4, ox = rem & 15;
  const int kh8 = (lane >> 4) << 3;

  f32x8 acc[8];
  #pragma unroll
  for (int nt = 0; nt < 8; ++nt) acc[nt] = zero8();

  #pragma unroll 1
  for (int t = 0; t < 256; ++t) {            // K-steps of 32 (same pixel, 32 ci)
    const int p = t >> 2, ky = p >> 3, kx = p & 7, ci0 = (t & 3) << 5;
    const bf16_t* ap =
        xb + ((size_t)(b * NN + (oy * 8 + ky) * 128 + ox * 8 + kx)) * CC + ci0 + kh8;
    bf16x16 a;
    #pragma unroll
    for (int j = 0; j < 8; ++j) { a[j] = ap[j]; a[8 + j] = ap[16 + j]; }
    __builtin_prefetch(ap + 4 * CC, 0, 1);
    bf16x16 bfr[8];
    #pragma unroll
    for (int nt = 0; nt < 8; ++nt)
      bfr[nt] = load_b(srw + (size_t)(nt * 16) * 8192, 8192, t * 32, lane);
    #pragma unroll
    for (int nt = 0; nt < 8; ++nt)
      acc[nt] = wmma_bf16(a, bfr[nt], acc[nt]);
  }
  #pragma unroll
  for (int nt = 0; nt < 8; ++nt) {
    const int col = nt * 16 + ccol(lane);
    #pragma unroll
    for (int i = 0; i < 8; ++i) {
      const int r = m0 + crow(lane, i);
      out[(size_t)r * CC + col] = acc[nt][i] + srb[col];
    }
  }
}

// ---------------------------------------------------------------------------
// LayerNorm over C=128, one 128-thread block per row, bf16 output.
// ---------------------------------------------------------------------------
__global__ void __launch_bounds__(128)
ln_kernel(const float* __restrict__ in, const float* __restrict__ g,
          const float* __restrict__ beta, bf16_t* __restrict__ out) {
  const int r = blockIdx.x, t = threadIdx.x;
  const float x = in[(size_t)r * CC + t];
  __shared__ float red[128];
  red[t] = x; __syncthreads();
  #pragma unroll
  for (int s = 64; s > 0; s >>= 1) { if (t < s) red[t] += red[t + s]; __syncthreads(); }
  const float mean = red[0] * (1.0f / CC);
  __syncthreads();
  const float d = x - mean;
  red[t] = d * d; __syncthreads();
  #pragma unroll
  for (int s = 64; s > 0; s >>= 1) { if (t < s) red[t] += red[t + s]; __syncthreads(); }
  const float var = red[0] * (1.0f / CC);
  out[(size_t)r * CC + t] = (bf16_t)(d * rsqrtf(var + 1e-5f) * g[t] + beta[t]);
}

// ---------------------------------------------------------------------------
// SR attention: q (B,N,2,64) x k (B,256,2,64) -> softmax(256) -> @ v
// grid = (256 tilegroups, HEADS, B); block = 128 (4 waves, 16 q-rows each).
// kv rows: kvb[(b*256+key)*256 + {0:k,128:v} + h*64 + d]
// ---------------------------------------------------------------------------
__global__ void __launch_bounds__(128)
attn_kernel(const bf16_t* __restrict__ qb, const bf16_t* __restrict__ kvb,
            float* __restrict__ oaf, bf16_t* __restrict__ oab) {
  const int b = blockIdx.z, h = blockIdx.y, tg = blockIdx.x;
  const int lane = threadIdx.x & 31, wave = threadIdx.x >> 5;

  __shared__ bf16_t k_lds[NKV * DH];           // 32 KB
  __shared__ bf16_t probs[4][16 * NKV];        // 32 KB (bf16 probs, row-major)

#if USE_ASYNC_LDS
  // stage K via gfx1250 async global->LDS DMA: 16B per lane per issue
  #pragma unroll
  for (int it = 0; it < 16; ++it) {
    const int c    = it * 128 + threadIdx.x;   // 2048 16B-chunks total
    const int key  = c >> 3, part = c & 7;
    const bf16_t* g = kvb + ((size_t)(b * NKV + key)) * 256 + h * DH + part * 8;
    __builtin_amdgcn_global_load_async_to_lds_b128(
        to_glob(g), to_lds(k_lds + key * DH + part * 8), 0, 0);
  }
  __builtin_amdgcn_s_wait_asynccnt(0);
#else
  for (int i = threadIdx.x; i < NKV * DH; i += 128) {
    const int key = i >> 6, d = i & 63;
    k_lds[i] = kvb[((size_t)(b * NKV + key)) * 256 + h * DH + d];
  }
#endif
  __syncthreads();

  const int m0 = tg * 64 + wave * 16;          // token tile within one image
  const bf16_t* qrow = qb + ((size_t)(b * NN + m0)) * CC + h * DH;
  const bf16x16 aq0 = load_a(qrow, CC, 0, lane);
  const bf16x16 aq1 = load_a(qrow, CC, 32, lane);

  f32x8 s[16];
  #pragma unroll
  for (int kt = 0; kt < 16; ++kt) s[kt] = zero8();
  #pragma unroll
  for (int kt = 0; kt < 16; ++kt) {            // fully unrolled: keeps s[] in VGPRs
    bf16x16 b0 = load_b(k_lds + (size_t)(kt * 16) * DH, DH, 0, lane);
    bf16x16 b1 = load_b(k_lds + (size_t)(kt * 16) * DH, DH, 32, lane);
    s[kt] = wmma_bf16(aq0, b0, s[kt]);
    s[kt] = wmma_bf16(aq1, b1, s[kt]);
  }

  const float scale = 0.125f;                  // 64^-0.5
  #pragma unroll
  for (int i = 0; i < 8; ++i) {
    float mx = -3.4e38f;
    #pragma unroll
    for (int kt = 0; kt < 16; ++kt) mx = fmaxf(mx, s[kt][i]);
    mx = half16_reduce_max(mx);
    float sum = 0.0f;
    #pragma unroll
    for (int kt = 0; kt < 16; ++kt) {
      const float e = __expf((s[kt][i] - mx) * scale);
      s[kt][i] = e; sum += e;
    }
    sum = half16_reduce_add(sum);
    const float inv = 1.0f / sum;
    #pragma unroll
    for (int kt = 0; kt < 16; ++kt) s[kt][i] *= inv;
  }

  bf16_t* pw = probs[wave];
  #pragma unroll
  for (int kt = 0; kt < 16; ++kt)
    #pragma unroll
    for (int i = 0; i < 8; ++i)
      pw[(size_t)crow(lane, i) * NKV + kt * 16 + ccol(lane)] = (bf16_t)s[kt][i];
  __syncthreads();

  f32x8 o[4];
  #pragma unroll
  for (int nt = 0; nt < 4; ++nt) o[nt] = zero8();
  const int khv = (lane >> 4) << 3;
  #pragma unroll 1
  for (int kt = 0; kt < 8; ++kt) {             // K = 256 keys, 32 per step
    bf16x16 ap = load_a(pw, NKV, kt * 32, lane);
    bf16x16 bvf[4];
    #pragma unroll
    for (int nt = 0; nt < 4; ++nt) {
      #pragma unroll
      for (int j = 0; j < 16; ++j)
        bvf[nt][j] = kvb[((size_t)(b * NKV + kt * 32 + khv + (j & 7) + ((j >> 3) << 4))) * 256 + CC +
                         h * DH + nt * 16 + ccol(lane)];
    }
    #pragma unroll
    for (int nt = 0; nt < 4; ++nt)
      o[nt] = wmma_bf16(ap, bvf[nt], o[nt]);
  }
  #pragma unroll
  for (int nt = 0; nt < 4; ++nt)
    #pragma unroll
    for (int i = 0; i < 8; ++i) {
      const size_t r = (size_t)b * NN + m0 + crow(lane, i);
      const int    c = h * DH + nt * 16 + ccol(lane);
      const float  v = o[nt][i];
      oaf[r * CC + c] = v;
      oab[r * CC + c] = (bf16_t)v;
    }
}

// ---------------------------------------------------------------------------
// Fused stage-B: judger -> 2-key MHA (8 heads, d=16) -> out proj -> residual
// -> final proj.  One wave owns a 16-token tile; all intermediates in LDS.
// ---------------------------------------------------------------------------
__global__ void __launch_bounds__(128)
stageB_kernel(const bf16_t* __restrict__ ab,  const float* __restrict__ af,
              const bf16_t* __restrict__ bb,
              const bf16_t* __restrict__ rj1, const float* __restrict__ rjb1,
              const bf16_t* __restrict__ rj2, const float* __restrict__ rjb2,
              const bf16_t* __restrict__ cain, const float* __restrict__ cainb,
              const bf16_t* __restrict__ caout, const float* __restrict__ caoutb,
              const bf16_t* __restrict__ projw, const float* __restrict__ projb,
              const float* __restrict__ knoise, const float* __restrict__ vnoise,
              float* __restrict__ outp) {
  const int lane = threadIdx.x & 31, wave = threadIdx.x >> 5;
  const size_t m0 = ((size_t)blockIdx.x * 4 + wave) * 16;

  __shared__ bf16_t lds[4][4][16 * CC];        // 4 waves x 4 bufs x 4KB = 64KB
  bf16_t* bufA = lds[wave][0];                 // h, later o
  bf16_t* bufB = lds[wave][1];                 // k0, later n
  bf16_t* bufC = lds[wave][2];                 // k1
  bf16_t* bufD = lds[wave][3];                 // v0

  const bf16_t* Arow = ab + m0 * CC;
  const bf16_t* Brow = bb + m0 * CC;

  // ---- judger stage 1: h = gelu([a,b] @ rj_w1^T + b1), rj_w1 (128 x 256)
  {
    f32x8 acc[8];
    #pragma unroll
    for (int nt = 0; nt < 8; ++nt) acc[nt] = zero8();
    #pragma unroll
    for (int ks = 0; ks < 4; ++ks) {
      bf16x16 a = load_a(Arow, CC, ks * 32, lane);
      bf16x16 bfr[8];
      #pragma unroll
      for (int nt = 0; nt < 8; ++nt)
        bfr[nt] = load_b(rj1 + (size_t)(nt * 16) * 256, 256, ks * 32, lane);
      #pragma unroll
      for (int nt = 0; nt < 8; ++nt) acc[nt] = wmma_bf16(a, bfr[nt], acc[nt]);
    }
    #pragma unroll
    for (int ks = 0; ks < 4; ++ks) {
      bf16x16 a = load_a(Brow, CC, ks * 32, lane);
      bf16x16 bfr[8];
      #pragma unroll
      for (int nt = 0; nt < 8; ++nt)
        bfr[nt] = load_b(rj1 + (size_t)(nt * 16) * 256, 256, 128 + ks * 32, lane);
      #pragma unroll
      for (int nt = 0; nt < 8; ++nt) acc[nt] = wmma_bf16(a, bfr[nt], acc[nt]);
    }
    #pragma unroll
    for (int nt = 0; nt < 8; ++nt) {
      const int col = nt * 16 + ccol(lane);
      #pragma unroll
      for (int i = 0; i < 8; ++i) {
        const float x = acc[nt][i] + rjb1[col];
        const float g = 0.5f * x * (1.0f + erff(x * 0.70710678118f)); // exact gelu
        bufA[(size_t)crow(lane, i) * CC + col] = (bf16_t)g;
      }
    }
  }
  __syncthreads();

  // ---- judger stage 2: s = softmax(h @ rj_w2^T + b2) over C, then build
  //      k0 = k_noise + a, k1 = a*s, v0 = v_noise + a (bf16 into LDS)
  {
    f32x8 sacc[8];
    #pragma unroll
    for (int nt = 0; nt < 8; ++nt) sacc[nt] = zero8();
    #pragma unroll
    for (int ks = 0; ks < 4; ++ks) {
      bf16x16 a = load_a(bufA, CC, ks * 32, lane);
      bf16x16 bfr[8];
      #pragma unroll
      for (int nt = 0; nt < 8; ++nt)
        bfr[nt] = load_b(rj2 + (size_t)(nt * 16) * CC, CC, ks * 32, lane);
      #pragma unroll
      for (int nt = 0; nt < 8; ++nt) sacc[nt] = wmma_bf16(a, bfr[nt], sacc[nt]);
    }
    #pragma unroll
    for (int nt = 0; nt < 8; ++nt) {
      const int col = nt * 16 + ccol(lane);
      #pragma unroll
      for (int i = 0; i < 8; ++i) sacc[nt][i] += rjb2[col];
    }
    #pragma unroll
    for (int i = 0; i < 8; ++i) {              // row softmax over 128 cols
      float mx = -3.4e38f;
      #pragma unroll
      for (int nt = 0; nt < 8; ++nt) mx = fmaxf(mx, sacc[nt][i]);
      mx = half16_reduce_max(mx);
      float sum = 0.0f;
      #pragma unroll
      for (int nt = 0; nt < 8; ++nt) { sacc[nt][i] = __expf(sacc[nt][i] - mx); sum += sacc[nt][i]; }
      sum = half16_reduce_add(sum);
      const float inv = 1.0f / sum;
      #pragma unroll
      for (int nt = 0; nt < 8; ++nt) sacc[nt][i] *= inv;
    }
    #pragma unroll
    for (int nt = 0; nt < 8; ++nt) {
      const int col = nt * 16 + ccol(lane);
      #pragma unroll
      for (int i = 0; i < 8; ++i) {
        const size_t r = m0 + crow(lane, i);
        const float av = af[r * CC + col];
        const size_t li = (size_t)crow(lane, i) * CC + col;
        bufB[li] = (bf16_t)(knoise[col] + av);
        bufC[li] = (bf16_t)(av * sacc[nt][i]);
        bufD[li] = (bf16_t)(vnoise[col] + av);
      }
    }
  }
  __syncthreads();

  // ---- 2-key MHA, one head per 16-wide N-tile (XHEADS=8, d=16)
  for (int h = 0; h < XHEADS; ++h) {
    f32x8 qp = zero8(), kp0 = zero8(), kp1 = zero8(), vp0 = zero8(), vp1 = zero8();
    const bf16_t* Wq_ = cain + (size_t)(0    + h * XD) * CC;
    const bf16_t* Wk_ = cain + (size_t)(CC   + h * XD) * CC;
    const bf16_t* Wv_ = cain + (size_t)(2*CC + h * XD) * CC;
    #pragma unroll
    for (int ks = 0; ks < 4; ++ks) {
      const bf16x16 bq = load_b(Wq_, CC, ks * 32, lane);
      const bf16x16 bk = load_b(Wk_, CC, ks * 32, lane);
      const bf16x16 bv = load_b(Wv_, CC, ks * 32, lane);
      const bf16x16 aa = load_a(Arow, CC, ks * 32, lane);
      const bf16x16 a0 = load_a(bufB, CC, ks * 32, lane);
      const bf16x16 a1 = load_a(bufC, CC, ks * 32, lane);
      const bf16x16 a2 = load_a(bufD, CC, ks * 32, lane);
      const bf16x16 a3 = load_a(Brow, CC, ks * 32, lane);
      qp  = wmma_bf16(aa, bq, qp);
      kp0 = wmma_bf16(a0, bk, kp0);
      kp1 = wmma_bf16(a1, bk, kp1);
      vp0 = wmma_bf16(a2, bv, vp0);
      vp1 = wmma_bf16(a3, bv, vp1);
    }
    const int col = h * XD + ccol(lane);
    const float bq_ = cainb[col], bk_ = cainb[CC + col], bv_ = cainb[2 * CC + col];
    #pragma unroll
    for (int i = 0; i < 8; ++i) {
      const float q  = qp[i]  + bq_;
      const float k0 = kp0[i] + bk_;
      const float k1 = kp1[i] + bk_;
      const float v0 = vp0[i] + bv_;
      const float v1 = vp1[i] + bv_;
      float s0 = half16_reduce_add(q * k0) * 0.25f;   // d=16 -> scale 0.25
      float s1 = half16_reduce_add(q * k1) * 0.25f;
      const float mx = fmaxf(s0, s1);
      const float e0 = __expf(s0 - mx), e1 = __expf(s1 - mx);
      const float ov = (e0 * v0 + e1 * v1) / (e0 + e1);
      bufA[(size_t)crow(lane, i) * CC + col] = (bf16_t)ov;  // o overwrites h
    }
  }
  __syncthreads();

  // ---- n = a + o @ out_w^T + out_b
  {
    f32x8 nacc[8];
    #pragma unroll
    for (int nt = 0; nt < 8; ++nt) nacc[nt] = zero8();
    #pragma unroll
    for (int ks = 0; ks < 4; ++ks) {
      bf16x16 a = load_a(bufA, CC, ks * 32, lane);
      bf16x16 bfr[8];
      #pragma unroll
      for (int nt = 0; nt < 8; ++nt)
        bfr[nt] = load_b(caout + (size_t)(nt * 16) * CC, CC, ks * 32, lane);
      #pragma unroll
      for (int nt = 0; nt < 8; ++nt) nacc[nt] = wmma_bf16(a, bfr[nt], nacc[nt]);
    }
    #pragma unroll
    for (int nt = 0; nt < 8; ++nt) {
      const int col = nt * 16 + ccol(lane);
      #pragma unroll
      for (int i = 0; i < 8; ++i) {
        const size_t r = m0 + crow(lane, i);
        const float nv = nacc[nt][i] + caoutb[col] + af[r * CC + col];
        bufB[(size_t)crow(lane, i) * CC + col] = (bf16_t)nv;   // n overwrites k0
      }
    }
  }
  __syncthreads();

  // ---- out = n @ proj_w^T + proj_b  (f32 to d_out)
  {
    f32x8 oacc[8];
    #pragma unroll
    for (int nt = 0; nt < 8; ++nt) oacc[nt] = zero8();
    #pragma unroll
    for (int ks = 0; ks < 4; ++ks) {
      bf16x16 a = load_a(bufB, CC, ks * 32, lane);
      bf16x16 bfr[8];
      #pragma unroll
      for (int nt = 0; nt < 8; ++nt)
        bfr[nt] = load_b(projw + (size_t)(nt * 16) * CC, CC, ks * 32, lane);
      #pragma unroll
      for (int nt = 0; nt < 8; ++nt) oacc[nt] = wmma_bf16(a, bfr[nt], oacc[nt]);
    }
    #pragma unroll
    for (int nt = 0; nt < 8; ++nt) {
      const int col = nt * 16 + ccol(lane);
      #pragma unroll
      for (int i = 0; i < 8; ++i) {
        const size_t r = m0 + crow(lane, i);
        outp[r * CC + col] = oacc[nt][i] + projb[col];
      }
    }
  }
}

// ---------------------------------------------------------------------------
// Host-side orchestration
// ---------------------------------------------------------------------------
extern "C" void kernel_launch(void* const* d_in, const int* in_sizes, int n_in,
                              void* d_out, int out_size, void* d_ws, size_t ws_size,
                              hipStream_t stream) {
  (void)in_sizes; (void)n_in; (void)out_size; (void)ws_size;
  const float* x0    = (const float*)d_in[0];
  const float* x1    = (const float*)d_in[1];
  const float* Wq    = (const float*)d_in[2];
  const float* bq    = (const float*)d_in[3];
  const float* Wkv   = (const float*)d_in[4];
  const float* bkv   = (const float*)d_in[5];
  const float* sr_w  = (const float*)d_in[6];
  const float* sr_b  = (const float*)d_in[7];
  const float* ln0g  = (const float*)d_in[8];
  const float* ln0b  = (const float*)d_in[9];
  const float* ln1g  = (const float*)d_in[10];
  const float* ln1b  = (const float*)d_in[11];
  const float* ca01w = (const float*)d_in[12];
  const float* ca01b = (const float*)d_in[13];
  const float* ca01ow= (const float*)d_in[14];
  const float* ca01ob= (const float*)d_in[15];
  const float* ca10w = (const float*)d_in[16];
  const float* ca10b = (const float*)d_in[17];
  const float* ca10ow= (const float*)d_in[18];
  const float* ca10ob= (const float*)d_in[19];
  const float* rjw1  = (const float*)d_in[20];
  const float* rjb1  = (const float*)d_in[21];
  const float* rjw2  = (const float*)d_in[22];
  const float* rjb2  = (const float*)d_in[23];
  const float* knoise= (const float*)d_in[24];
  const float* vnoise= (const float*)d_in[25];
  const float* projw = (const float*)d_in[26];
  const float* projb = (const float*)d_in[27];
  float* out = (float*)d_out;

  // workspace carve-out (~172 MB)
  char* w = (char*)d_ws;
  auto alloc = [&](size_t bytes) { char* p = w; w += (bytes + 255) & ~(size_t)255; return p; };
  const size_t MC = (size_t)MTOT * CC;
  bf16_t* x0b   = (bf16_t*)alloc(MC * 2);
  bf16_t* x1b   = (bf16_t*)alloc(MC * 2);
  bf16_t* q0b   = (bf16_t*)alloc(MC * 2);
  bf16_t* q1b   = (bf16_t*)alloc(MC * 2);
  bf16_t* x0ab  = (bf16_t*)alloc(MC * 2);
  bf16_t* x1ab  = (bf16_t*)alloc(MC * 2);
  float*  x0af  = (float*) alloc(MC * 4);
  float*  x1af  = (float*) alloc(MC * 4);
  float*  convf = (float*) alloc((size_t)BB * NKV * CC * 4);
  bf16_t* xiln  = (bf16_t*)alloc((size_t)BB * NKV * CC * 2);
  bf16_t* kv0b  = (bf16_t*)alloc((size_t)BB * NKV * 2 * CC * 2);
  bf16_t* kv1b  = (bf16_t*)alloc((size_t)BB * NKV * 2 * CC * 2);
  bf16_t* wqb   = (bf16_t*)alloc(CC * CC * 2);
  bf16_t* wkvb  = (bf16_t*)alloc(2 * CC * CC * 2);
  bf16_t* srwb  = (bf16_t*)alloc((size_t)CC * CC * SRR * SRR * 2);
  bf16_t* rj1b  = (bf16_t*)alloc(CC * 2 * CC * 2);
  bf16_t* rj2b  = (bf16_t*)alloc(CC * CC * 2);
  bf16_t* c01b  = (bf16_t*)alloc(3 * CC * CC * 2);
  bf16_t* c01ob = (bf16_t*)alloc(CC * CC * 2);
  bf16_t* c10b  = (bf16_t*)alloc(3 * CC * CC * 2);
  bf16_t* c10ob = (bf16_t*)alloc(CC * CC * 2);
  bf16_t* prjb  = (bf16_t*)alloc(CC * CC * 2);

  auto cvt = [&](const float* s, bf16_t* d, int n) {
    cvt_f32_bf16<<<(n + 255) / 256, 256, 0, stream>>>(s, d, n);
  };
  cvt(x0, x0b, MTOT * CC);   cvt(x1, x1b, MTOT * CC);
  cvt(Wq, wqb, CC * CC);     cvt(Wkv, wkvb, 2 * CC * CC);
  cvt(rjw1, rj1b, CC * 2 * CC); cvt(rjw2, rj2b, CC * CC);
  cvt(ca01w, c01b, 3 * CC * CC); cvt(ca01ow, c01ob, CC * CC);
  cvt(ca10w, c10b, 3 * CC * CC); cvt(ca10ow, c10ob, CC * CC);
  cvt(projw, prjb, CC * CC);
  repack_srw<<<(CC * CC * SRR * SRR + 255) / 256, 256, 0, stream>>>(sr_w, srwb);

  // q projections (M = 65536, K = 128, N = 128)
  gemm_bias_bf16<8, 4><<<MTOT / 64, 128, 0, stream>>>(x0b, wqb, bq, q0b, MTOT, CC);
  gemm_bias_bf16<8, 4><<<MTOT / 64, 128, 0, stream>>>(x1b, wqb, bq, q1b, MTOT, CC);

  // stream 0: conv -> LN -> kv
  conv_kernel<<<BB * NKV / 64, 128, 0, stream>>>(x0b, srwb, sr_b, convf);
  ln_kernel<<<BB * NKV, 128, 0, stream>>>(convf, ln0g, ln0b, xiln);
  gemm_bias_bf16<16, 4><<<BB * NKV / 64, 128, 0, stream>>>(xiln, wkvb, bkv, kv0b, BB * NKV, 2 * CC);
  // stream 1
  conv_kernel<<<BB * NKV / 64, 128, 0, stream>>>(x1b, srwb, sr_b, convf);
  ln_kernel<<<BB * NKV, 128, 0, stream>>>(convf, ln1g, ln1b, xiln);
  gemm_bias_bf16<16, 4><<<BB * NKV / 64, 128, 0, stream>>>(xiln, wkvb, bkv, kv1b, BB * NKV, 2 * CC);

  // SR attention -> x0a / x1a (f32 + bf16)
  attn_kernel<<<dim3(NN / 64, HEADS, BB), 128, 0, stream>>>(q0b, kv0b, x0af, x0ab);
  attn_kernel<<<dim3(NN / 64, HEADS, BB), 128, 0, stream>>>(q1b, kv1b, x1af, x1ab);

  // fused stage B (judger + mha2 + out proj + residual + final proj)
  stageB_kernel<<<MTOT / 64, 128, 0, stream>>>(
      x0ab, x0af, x1ab, rj1b, rjb1, rj2b, rjb2,
      c01b, ca01b, c01ob, ca01ob, prjb, projb,
      knoise + 0 * CC, vnoise + 0 * CC, out);
  stageB_kernel<<<MTOT / 64, 128, 0, stream>>>(
      x1ab, x1af, x0ab, rj1b, rjb1, rj2b, rjb2,
      c10b, ca10b, c10ob, ca10ob, prjb, projb,
      knoise + 1 * CC, vnoise + 1 * CC, out + MC);
}
